// SelfAttention_54133767799243
// MI455X (gfx1250) — matrix operand order repaired and checked by reading, not compiled, not run on hardware
//
#include <hip/hip_runtime.h>

#define B_ 2
#define T_ 4096
#define C_ 512
#define H_ 8
#define D_ 64
#define M_ (B_*T_)          // 8192 rows
#define NQKV_ (3*C_)        // 1536

typedef __attribute__((ext_vector_type(16))) _Float16 v16h;
typedef __attribute__((ext_vector_type(8)))  float    v8f;
typedef __attribute__((ext_vector_type(4)))  unsigned int u32x4;

union F16Frag { v16h v; u32x4 q[2]; _Float16 h[16]; unsigned int u[8]; };

__device__ __forceinline__ v8f wmma16(v16h a, v16h b, v8f c) {
  // D = A(16x32 f16) * B(32x16 f16) + C(16x16 f32)
  return __builtin_amdgcn_wmma_f32_16x16x32_f16(false, a, false, b, (short)0, c, false, false);
}

// ---------------------------------------------------------------- conversions
__global__ void cvt_f32_f16(const float* __restrict__ in, _Float16* __restrict__ out, int n) {
  int i = blockIdx.x * blockDim.x + threadIdx.x;
  if (i < n) out[i] = (_Float16)in[i];
}

// in[K][N] f32 -> out[N][K] f16 (weights transposed once so GEMM B-fragments
// become contiguous k-runs readable with global_load_b128)
__global__ void cvt_f32_f16_t(const float* __restrict__ in, _Float16* __restrict__ out,
                              int K, int N) {
  int i = blockIdx.x * blockDim.x + threadIdx.x;
  if (i < K * N) {
    int k = i / N, n = i - k * N;
    out[(size_t)n * K + k] = (_Float16)in[i];
  }
}

// ---------------------------------------------------------------- QKV GEMM (no LDS)
// X[M_,C_] f16 @ Wt[NQKV_,C_] (pre-transposed) + bias.
// q,k -> [B*H, T, D];  v -> [B*H, D, T] (transposed for the PV matmul).
// Softmax scale * log2(e) folded into Q.
__global__ __launch_bounds__(256)
void gemm_qkv(const _Float16* __restrict__ X, const _Float16* __restrict__ Wt,
              const float* __restrict__ bias,
              _Float16* __restrict__ qh, _Float16* __restrict__ kh, _Float16* __restrict__ vt)
{
  const int tid = threadIdx.x;
  const int lane = tid & 31, w = tid >> 5;
  const int ln16 = lane & 15, hv = lane >> 4;
  const int wm = w >> 2, wn = w & 3;               // 2 x 4 wave grid
  const int m0 = blockIdx.x * 64;
  const int n0 = blockIdx.y * 128;

  const _Float16* arow[2];
  const _Float16* bcol[2];
  #pragma unroll
  for (int mt = 0; mt < 2; mt++)
    arow[mt] = X + (size_t)(m0 + wm * 32 + mt * 16 + ln16) * C_;
  #pragma unroll
  for (int nt = 0; nt < 2; nt++)
    bcol[nt] = Wt + (size_t)(n0 + wn * 32 + nt * 16 + ln16) * C_;

  v8f zero = {};
  v8f acc[2][2] = {{zero, zero}, {zero, zero}};

  for (int k0 = 0; k0 < C_; k0 += 32) {
    F16Frag a[2], b[2];
    #pragma unroll
    for (int mt = 0; mt < 2; mt++) {
      a[mt].q[0] = *(const u32x4*)(arow[mt] + k0 + 8 * hv);
      a[mt].q[1] = *(const u32x4*)(arow[mt] + k0 + 16 + 8 * hv);
    }
    #pragma unroll
    for (int nt = 0; nt < 2; nt++) {
      b[nt].q[0] = *(const u32x4*)(bcol[nt] + k0 + 8 * hv);
      b[nt].q[1] = *(const u32x4*)(bcol[nt] + k0 + 16 + 8 * hv);
    }
    #pragma unroll
    for (int mt = 0; mt < 2; mt++)
      #pragma unroll
      for (int nt = 0; nt < 2; nt++)
        acc[mt][nt] = wmma16(a[mt].v, b[nt].v, acc[mt][nt]);
  }

  const float qscale = 0.18033688011112042f; // (1/sqrt(64)) * log2(e)
  #pragma unroll
  for (int mt = 0; mt < 2; mt++)
    #pragma unroll
    for (int nt = 0; nt < 2; nt++) {
      int ng = n0 + wn * 32 + nt * 16 + ln16;
      float bv = bias[ng];
      int which = ng >> 9;            // 0=q 1=k 2=v
      int cc = ng & 511;
      int hh = cc >> 6, dd = cc & 63;
      float mul = (which == 0) ? qscale : 1.0f;
      #pragma unroll
      for (int r = 0; r < 8; r++) {
        int mg = m0 + wm * 32 + mt * 16 + r + 8 * hv;
        int bb = mg >> 12, tt = mg & (T_ - 1);
        float v = (acc[mt][nt][r] + bv) * mul;
        int bhh = bb * H_ + hh;
        if (which == 0)      qh[(size_t)(bhh * T_ + tt) * D_ + dd] = (_Float16)v;
        else if (which == 1) kh[(size_t)(bhh * T_ + tt) * D_ + dd] = (_Float16)v;
        else                 vt[(size_t)(bhh * D_ + dd) * T_ + tt] = (_Float16)v;
      }
    }
}

// ---------------------------------------------------------------- flash attention
// Per wave: 32 query rows (two 16-row tiles sharing every K / V^T fragment),
// fully independent (no LDS, no barriers).
// S^T = K*Q^T, online softmax per q-tile, P^T->B-fragment via shfl_xor(16),
// O^T += V^T * P^T with V^T fragments from the transposed-V layout.
__global__ __launch_bounds__(256)
void flash_attn(const _Float16* __restrict__ qh, const _Float16* __restrict__ kh,
                const _Float16* __restrict__ vt, _Float16* __restrict__ y16)
{
  const int tid = threadIdx.x;
  const int lane = tid & 31, w = tid >> 5;
  const int ln16 = lane & 15, hv = lane >> 4;
  const int bh = blockIdx.y;
  const int b = bh >> 3, hh = bh & 7;
  const int qrow0 = blockIdx.x * 256 + w * 32 + ln16;   // q-tile qt adds 16

  const _Float16* Qp  = qh + (size_t)bh * T_ * D_;
  const _Float16* Kp  = kh + (size_t)bh * T_ * D_;
  const _Float16* Vtp = vt + (size_t)bh * D_ * T_;

  F16Frag qf[2][2];                    // [qt][ks] Q^T B-fragments, loop invariant
  #pragma unroll
  for (int qt = 0; qt < 2; qt++)
    #pragma unroll
    for (int ks = 0; ks < 2; ks++) {
      const _Float16* p = Qp + (size_t)(qrow0 + qt * 16) * D_ + ks * 32 + hv * 8;
      qf[qt][ks].q[0] = *(const u32x4*)p;
      qf[qt][ks].q[1] = *(const u32x4*)(p + 16);
    }

  v8f zero = {};
  v8f accT[2][4] = {{zero, zero, zero, zero}, {zero, zero, zero, zero}};
  float mrow[2] = {-1e30f, -1e30f};
  float lrow[2] = {0.0f, 0.0f};
  F16Frag pb[2];

  for (int kt = 0; kt < T_; kt += 32) {
    if (kt + 32 < T_) {   // cover next tile's latency
      __builtin_prefetch(Kp + (size_t)(kt + 32 + ln16) * D_, 0, 3);
      __builtin_prefetch(Vtp + (size_t)(ln16 * 4 + hv) * T_ + kt + 32, 0, 3);
    }

    // K A-fragments for this key tile, shared by both q-tiles
    F16Frag a[2][2];   // [mt][ks]
    #pragma unroll
    for (int mt = 0; mt < 2; mt++)
      #pragma unroll
      for (int ks = 0; ks < 2; ks++) {
        const _Float16* kp = Kp + (size_t)(kt + mt * 16 + ln16) * D_ + ks * 32;
        a[mt][ks].q[0] = *(const u32x4*)(kp + 8 * hv);
        a[mt][ks].q[1] = *(const u32x4*)(kp + 16 + 8 * hv);
      }

    #pragma unroll
    for (int qt = 0; qt < 2; qt++) {
      // S^T = K * Q^T : two 16-key sub-tiles
      v8f s[2];
      float tmax = -1e30f;
      #pragma unroll
      for (int mt = 0; mt < 2; mt++) {
        v8f c = zero;
        #pragma unroll
        for (int ks = 0; ks < 2; ks++)
          c = wmma16(a[mt][ks].v, qf[qt][ks].v, c);
        s[mt] = c;
        #pragma unroll
        for (int r = 0; r < 8; r++) tmax = fmaxf(tmax, c[r]);
      }
      tmax = fmaxf(tmax, __shfl_xor(tmax, 16, 32));
      float mnew = fmaxf(mrow[qt], tmax);
      float alpha = exp2f(mrow[qt] - mnew);   // scores already in log2 domain
      float rsum = 0.0f;
      F16Frag own;                            // h[0..7]=p(mt0), h[8..15]=p(mt1)
      #pragma unroll
      for (int mt = 0; mt < 2; mt++)
        #pragma unroll
        for (int r = 0; r < 8; r++) {
          float pv = exp2f(s[mt][r] - mnew);
          rsum += pv;
          own.h[mt * 8 + r] = (_Float16)pv;
        }
      rsum += __shfl_xor(rsum, 16, 32);
      lrow[qt] = lrow[qt] * alpha + rsum;
      mrow[qt] = mnew;
      #pragma unroll
      for (int dt = 0; dt < 4; dt++)
        #pragma unroll
        for (int r = 0; r < 8; r++) accT[qt][dt][r] *= alpha;

      // Build P^T B-fragment: lane(hv,n) needs keys 16*hv..16*hv+15 of column q=n;
      // partner (lane^16) owns the other half of each 16-key run.
      #pragma unroll
      for (int i = 0; i < 4; i++) {
        unsigned send = hv ? own.u[i] : own.u[4 + i];       // what partner needs
        unsigned rec  = (unsigned)__shfl_xor((int)send, 16, 32);
        pb[qt].u[i]     = hv ? rec : own.u[i];
        pb[qt].u[4 + i] = hv ? own.u[4 + i] : rec;
      }
    }

    #pragma unroll
    for (int dt = 0; dt < 4; dt++) {   // O^T += V^T * P^T, V frag shared by q-tiles
      F16Frag va;
      const _Float16* vp = Vtp + (size_t)(dt * 16 + ln16) * T_ + kt;
      va.q[0] = *(const u32x4*)(vp + 16 * hv);
      va.q[1] = *(const u32x4*)(vp + 16 * hv + 8);
      #pragma unroll
      for (int qt = 0; qt < 2; qt++)
        accT[qt][dt] = wmma16(va.v, pb[qt].v, accT[qt][dt]);
    }
  }

  #pragma unroll
  for (int qt = 0; qt < 2; qt++) {
    float invl = 1.0f / lrow[qt];
    _Float16* yrow = y16 + (size_t)(b * T_ + qrow0 + qt * 16) * C_ + hh * D_;
    #pragma unroll
    for (int dt = 0; dt < 4; dt++) {
      F16Frag o;
      #pragma unroll
      for (int r = 0; r < 8; r++) o.h[r] = (_Float16)(accT[qt][dt][r] * invl);
      *(u32x4*)(yrow + dt * 16 + 8 * hv) = o.q[0];
    }
  }
}

// ---------------------------------------------------------------- projection GEMM (no LDS)
// Y[M_,C_] f16 @ Wpt[C_,C_] (pre-transposed) + bias -> out f32 [M_,C_]
__global__ __launch_bounds__(256)
void gemm_proj(const _Float16* __restrict__ X, const _Float16* __restrict__ Wt,
               const float* __restrict__ bias, float* __restrict__ out)
{
  const int tid = threadIdx.x;
  const int lane = tid & 31, w = tid >> 5;
  const int ln16 = lane & 15, hv = lane >> 4;
  const int wm = w >> 2, wn = w & 3;
  const int m0 = blockIdx.x * 64;
  const int n0 = blockIdx.y * 128;

  const _Float16* arow[2];
  const _Float16* bcol[2];
  #pragma unroll
  for (int mt = 0; mt < 2; mt++)
    arow[mt] = X + (size_t)(m0 + wm * 32 + mt * 16 + ln16) * C_;
  #pragma unroll
  for (int nt = 0; nt < 2; nt++)
    bcol[nt] = Wt + (size_t)(n0 + wn * 32 + nt * 16 + ln16) * C_;

  v8f zero = {};
  v8f acc[2][2] = {{zero, zero}, {zero, zero}};

  for (int k0 = 0; k0 < C_; k0 += 32) {
    F16Frag a[2], b[2];
    #pragma unroll
    for (int mt = 0; mt < 2; mt++) {
      a[mt].q[0] = *(const u32x4*)(arow[mt] + k0 + 8 * hv);
      a[mt].q[1] = *(const u32x4*)(arow[mt] + k0 + 16 + 8 * hv);
    }
    #pragma unroll
    for (int nt = 0; nt < 2; nt++) {
      b[nt].q[0] = *(const u32x4*)(bcol[nt] + k0 + 8 * hv);
      b[nt].q[1] = *(const u32x4*)(bcol[nt] + k0 + 16 + 8 * hv);
    }
    #pragma unroll
    for (int mt = 0; mt < 2; mt++)
      #pragma unroll
      for (int nt = 0; nt < 2; nt++)
        acc[mt][nt] = wmma16(a[mt].v, b[nt].v, acc[mt][nt]);
  }

  #pragma unroll
  for (int mt = 0; mt < 2; mt++)
    #pragma unroll
    for (int nt = 0; nt < 2; nt++) {
      int ng = n0 + wn * 32 + nt * 16 + ln16;
      float bv = bias[ng];
      #pragma unroll
      for (int r = 0; r < 8; r++) {
        int mg = m0 + wm * 32 + mt * 16 + r + 8 * hv;
        out[(size_t)mg * C_ + ng] = acc[mt][nt][r] + bv;
      }
    }
}

// ---------------------------------------------------------------- launcher
extern "C" void kernel_launch(void* const* d_in, const int* in_sizes, int n_in,
                              void* d_out, int out_size, void* d_ws, size_t ws_size,
                              hipStream_t stream) {
  const float* x      = (const float*)d_in[0];
  const float* W_attn = (const float*)d_in[1];
  const float* b_attn = (const float*)d_in[2];
  const float* W_proj = (const float*)d_in[3];
  const float* b_proj = (const float*)d_in[4];
  float* out = (float*)d_out;

  size_t off = 0;
  auto carve = [&](size_t bytes) {
    void* p = (char*)d_ws + off;
    off += (bytes + 255) & ~(size_t)255;
    return p;
  };
  _Float16* x16  = (_Float16*)carve((size_t)M_ * C_ * 2);
  _Float16* Wa16 = (_Float16*)carve((size_t)C_ * NQKV_ * 2);   // [NQKV_][C_]
  _Float16* Wp16 = (_Float16*)carve((size_t)C_ * C_ * 2);      // [C_][C_] transposed
  _Float16* qh   = (_Float16*)carve((size_t)B_ * H_ * T_ * D_ * 2);
  _Float16* kh   = (_Float16*)carve((size_t)B_ * H_ * T_ * D_ * 2);
  _Float16* vt   = (_Float16*)carve((size_t)B_ * H_ * D_ * T_ * 2);
  _Float16* y16  = (_Float16*)carve((size_t)M_ * C_ * 2);

  int nx = M_ * C_, nwa = C_ * NQKV_, nwp = C_ * C_;
  cvt_f32_f16<<<(nx + 255) / 256, 256, 0, stream>>>(x, x16, nx);
  cvt_f32_f16_t<<<(nwa + 255) / 256, 256, 0, stream>>>(W_attn, Wa16, C_, NQKV_);
  cvt_f32_f16_t<<<(nwp + 255) / 256, 256, 0, stream>>>(W_proj, Wp16, C_, C_);

  gemm_qkv<<<dim3(M_ / 64, NQKV_ / 128), 256, 0, stream>>>(x16, Wa16, b_attn, qh, kh, vt);
  flash_attn<<<dim3(T_ / 256, B_ * H_), 256, 0, stream>>>(qh, kh, vt, y16);
  gemm_proj<<<dim3(M_ / 64, C_ / 128), 256, 0, stream>>>(y16, Wp16, b_proj, out);
}
